// MambaBlock_24369644437669
// MI455X (gfx1250) — matrix operand order repaired and checked by reading, not compiled, not run on hardware
//
#include <hip/hip_runtime.h>
#include <hip/hip_bf16.h>

typedef __attribute__((ext_vector_type(16))) __bf16 v16bf;
typedef __attribute__((ext_vector_type(8)))  float  v8f;

#ifndef __has_builtin
#define __has_builtin(x) 0
#endif

#if __has_builtin(__builtin_amdgcn_global_load_async_to_lds_b128)
#define HAVE_ASYNC_LDS 1
#else
#define HAVE_ASYNC_LDS 0
#endif
#if __has_builtin(__builtin_amdgcn_s_wait_asynccnt)
#define HAVE_WAIT_ASYNC 1
#else
#define HAVE_WAIT_ASYNC 0
#endif

typedef int v4i __attribute__((vector_size(16)));
typedef __attribute__((address_space(1))) v4i as1_v4i;
typedef __attribute__((address_space(3))) v4i as3_v4i;

// 16B global -> LDS copy: async (ASYNCcnt) when available, sync fallback.
__device__ __forceinline__ void cp16(const __bf16* g, __bf16* l) {
#if HAVE_ASYNC_LDS
    __builtin_amdgcn_global_load_async_to_lds_b128((as1_v4i*)g, (as3_v4i*)l, 0, 0);
#else
    *reinterpret_cast<uint4*>(l) = *reinterpret_cast<const uint4*>(g);
#endif
}

template <int N>
__device__ __forceinline__ void wait_async() {
#if HAVE_WAIT_ASYNC
    __builtin_amdgcn_s_wait_asynccnt(N);
#elif HAVE_ASYNC_LDS
    asm volatile("s_wait_asynccnt %0" ::"i"(N) : "memory");
#endif
}

// ---------------------------------------------------------------------------
// f32 -> bf16 conversion (weights / operand staging)
// ---------------------------------------------------------------------------
__global__ __launch_bounds__(256) void f32_to_bf16_kernel(const float* __restrict__ src,
                                                          __bf16* __restrict__ dst,
                                                          size_t n) {
    size_t i = (size_t)blockIdx.x * 256 + threadIdx.x;
    if (i < n) dst[i] = (__bf16)src[i];
}

// ---------------------------------------------------------------------------
// LayerNorm over d_model, emits bf16 A-operand for the in-projection GEMM
// ---------------------------------------------------------------------------
__global__ __launch_bounds__(256) void ln_kernel(const float* __restrict__ x,
                                                 const float* __restrict__ g,
                                                 const float* __restrict__ b,
                                                 __bf16* __restrict__ out, int dm) {
    int row = blockIdx.x;
    int tid = threadIdx.x;
    const float* xr = x + (size_t)row * dm;
    float s = 0.f, ss = 0.f;
    for (int i = tid; i < dm; i += 256) { float v = xr[i]; s += v; ss += v * v; }
    for (int off = 16; off > 0; off >>= 1) {
        s  += __shfl_down(s,  off);
        ss += __shfl_down(ss, off);
    }
    __shared__ float rs[8], rss[8];
    __shared__ float mu_s, rstd_s;
    if ((tid & 31) == 0) { rs[tid >> 5] = s; rss[tid >> 5] = ss; }
    __syncthreads();
    if (tid == 0) {
        float S = 0.f, SS = 0.f;
        for (int i = 0; i < 8; ++i) { S += rs[i]; SS += rss[i]; }
        float mu  = S / dm;
        float var = SS / dm - mu * mu;
        mu_s = mu; rstd_s = rsqrtf(var + 1e-5f);
    }
    __syncthreads();
    float mu = mu_s, r = rstd_s;
    __bf16* orow = out + (size_t)row * dm;
    for (int i = tid; i < dm; i += 256)
        orow[i] = (__bf16)(((xr[i] - mu) * r) * g[i] + b[i]);
}

// ---------------------------------------------------------------------------
// bf16 WMMA GEMM:  C[M,N] = A[M,K] * W[N,K]^T (+ Res)
// 128x128x32 block tile, 8 waves, each wave = 2x4 grid of 16x16x32 WMMAs.
// Tiles stream global->LDS via GLOBAL_LOAD_ASYNC_TO_LDS_B128 with double
// buffering; the A-tile K-permutation ([0..7,16..23,8..15,24..31]) is applied
// per 16B async block so each lane's WMMA fragment is one contiguous 32B
// ds_load matching the ISA 16-bit A/B VGPR layouts.
// ---------------------------------------------------------------------------
#define BM 128
#define BN 128
#define BK 32
#define LDT 48                 // padded LDS row pitch in bf16 (96B, 32B aligned)
#define ATILE (BM * LDT)

__global__ __launch_bounds__(256)
void gemm_bf16_wmma(const __bf16* __restrict__ A, const __bf16* __restrict__ W,
                    const float* __restrict__ Res, float* __restrict__ Cout,
                    int M, int N, int K) {
    __shared__ __attribute__((aligned(32))) __bf16 As[2][ATILE];
    __shared__ __attribute__((aligned(32))) __bf16 Bs[2][ATILE];

    int tid  = threadIdx.x;
    int lane = tid & 31;
    int wid  = tid >> 5;
    int wm   = (wid & 3) * 32;   // wave M offset in tile
    int wn   = (wid >> 2) * 64;  // wave N offset in tile
    int bm   = blockIdx.y * BM;
    int bn   = blockIdx.x * BN;

    v8f acc[2][4];
    #pragma unroll
    for (int mi = 0; mi < 2; ++mi)
        #pragma unroll
        for (int ni = 0; ni < 4; ++ni)
            #pragma unroll
            for (int v = 0; v < 8; ++v) acc[mi][ni][v] = 0.f;

    int lr = tid >> 1;           // 0..127 : row within tile
    int lk = (tid & 1) * 16;     // 0 or 16 : K half
    int d0 = lk ? 8 : 0;         // permuted A placement (per-16B block)
    int d1 = lk ? 24 : 16;
    int fl = lane & 15;          // fragment row (M for A, N for B)
    int fo = (lane >> 4) * 16;   // fragment 32B half select

    const __bf16* gA = A + (size_t)(bm + lr) * K + lk;
    const __bf16* gW = W + (size_t)(bn + lr) * K + lk;

    auto issue_tile = [&](int t, int buf) {
        const __bf16* ga = gA + t * BK;
        __bf16* la = &As[buf][lr * LDT];
        cp16(ga,     la + d0);       // K[0..7]   / K[16..23]
        cp16(ga + 8, la + d1);       // K[8..15]  / K[24..31]
        const __bf16* gb = gW + t * BK;
        __bf16* lb = &Bs[buf][lr * LDT];
        cp16(gb,     lb + d0);
        cp16(gb + 8, lb + d1);
    };

    const int NT = K / BK;
    issue_tile(0, 0);
    for (int t = 0; t < NT; ++t) {
        int buf = t & 1;
        if (t + 1 < NT) {
            issue_tile(t + 1, buf ^ 1);   // prefetch next tile into other buffer
            wait_async<4>();              // oldest 4 (current tile) complete
        } else {
            wait_async<0>();
        }
        __syncthreads();

        v16bf afrag[2], bfrag[4];
        #pragma unroll
        for (int mi = 0; mi < 2; ++mi)
            afrag[mi] = *reinterpret_cast<const v16bf*>(&As[buf][(wm + mi * 16 + fl) * LDT + fo]);
        #pragma unroll
        for (int ni = 0; ni < 4; ++ni)
            bfrag[ni] = *reinterpret_cast<const v16bf*>(&Bs[buf][(wn + ni * 16 + fl) * LDT + fo]);

        #pragma unroll
        for (int mi = 0; mi < 2; ++mi)
            #pragma unroll
            for (int ni = 0; ni < 4; ++ni)
                acc[mi][ni] = __builtin_amdgcn_wmma_f32_16x16x32_bf16(
                    false, afrag[mi], false, bfrag[ni],
                    (short)0, acc[mi][ni], false, false);
        __syncthreads();   // done reading buf before it is overwritten
    }

    #pragma unroll
    for (int mi = 0; mi < 2; ++mi) {
        #pragma unroll
        for (int v = 0; v < 8; ++v) {
            int r = bm + wm + mi * 16 + ((lane < 16) ? v : v + 8);
            #pragma unroll
            for (int ni = 0; ni < 4; ++ni) {
                int c = bn + wn + ni * 16 + fl;
                size_t o = (size_t)r * N + c;
                float val = acc[mi][ni][v];
                if (Res) val += Res[o];
                Cout[o] = val;
            }
        }
    }
}

// ---------------------------------------------------------------------------
// depthwise causal conv (k=4) + SiLU -> u
// ---------------------------------------------------------------------------
__global__ __launch_bounds__(256) void conv_silu_kernel(const float* __restrict__ xz,
                                                        const float* __restrict__ cw,
                                                        const float* __restrict__ cb,
                                                        float* __restrict__ u,
                                                        int L, int din, int rowstride,
                                                        size_t total) {
    size_t idx = (size_t)blockIdx.x * 256 + threadIdx.x;
    if (idx >= total) return;
    int c = (int)(idx % din);
    size_t bt = idx / din;
    int t = (int)(bt % L);
    size_t brow = bt - t;  // b*L
    float acc = cb[c];
    #pragma unroll
    for (int k = 0; k < 4; ++k) {
        int tt = t - 3 + k;
        if (tt >= 0) acc += cw[c * 4 + k] * xz[(brow + tt) * (size_t)rowstride + c];
    }
    float sig = 1.f / (1.f + __expf(-acc));
    u[idx] = acc * sig;
}

// ---------------------------------------------------------------------------
// dbc = u @ xproj_w^T   (N=96, K=2048)  -- small-N GEMV per row
// ---------------------------------------------------------------------------
__global__ __launch_bounds__(128) void xproj_kernel(const float* __restrict__ u,
                                                    const float* __restrict__ w,
                                                    float* __restrict__ dbc,
                                                    int din, int P) {
    int row = blockIdx.x;
    __shared__ float us[2048];
    for (int i = threadIdx.x; i < din; i += 128) us[i] = u[(size_t)row * din + i];
    __syncthreads();
    int j = threadIdx.x;
    if (j < P) {
        const float* wr = w + (size_t)j * din;
        float acc = 0.f;
        for (int k = 0; k < din; ++k) acc = fmaf(wr[k], us[k], acc);
        dbc[(size_t)row * P + j] = acc;
    }
}

// ---------------------------------------------------------------------------
// delta = softplus(delta_raw @ dt_w^T + dt_b)   (K=64, N=2048)
// ---------------------------------------------------------------------------
__global__ __launch_bounds__(256) void delta_kernel(const float* __restrict__ dbc,
                                                    const float* __restrict__ dt_w,
                                                    const float* __restrict__ dt_b,
                                                    float* __restrict__ delta,
                                                    int din, int R, int P) {
    int row = blockIdx.x;
    __shared__ float dr[64];
    if (threadIdx.x < R) dr[threadIdx.x] = dbc[(size_t)row * P + threadIdx.x];
    __syncthreads();
    for (int n = threadIdx.x; n < din; n += 256) {
        const float* wr = dt_w + (size_t)n * R;
        float acc = dt_b[n];
        #pragma unroll 8
        for (int r = 0; r < 64; ++r) acc = fmaf(wr[r], dr[r], acc);
        float sp = (acc > 20.f) ? acc : log1pf(__expf(acc));
        delta[(size_t)row * din + n] = sp;
    }
}

// ---------------------------------------------------------------------------
// selective scan, fused with +u*D and *silu(z); emits bf16 y for out-GEMM.
// One thread per (batch, channel), 16 states in registers. B_t/C_t are
// broadcast wave-locally with __shfl (no LDS, no barriers on the 2048-step
// sequential path).
// ---------------------------------------------------------------------------
__global__ __launch_bounds__(256) void scan_kernel(const float* __restrict__ delta,
                                                   const float* __restrict__ u,
                                                   const float* __restrict__ dbc,
                                                   const float* __restrict__ xz,
                                                   const float* __restrict__ A_log,
                                                   const float* __restrict__ Dv,
                                                   __bf16* __restrict__ ybf,
                                                   int L, int din, int P, int R,
                                                   int xzstride) {
    int b    = blockIdx.x >> 3;
    int ch   = (blockIdx.x & 7) * 256 + threadIdx.x;
    int lane = threadIdx.x & 31;

    float Arow[16], h[16];
    #pragma unroll
    for (int n = 0; n < 16; ++n) {
        Arow[n] = -__expf(A_log[(size_t)ch * 16 + n]);
        h[n] = 0.f;
    }
    float Dch = Dv[ch];

    for (int t = 0; t < L; ++t) {
        size_t bt = (size_t)b * L + t;
        float bc = dbc[bt * P + R + lane];   // lane 0..15 = B_t, 16..31 = C_t
        float d  = delta[bt * din + ch];
        float uu = u[bt * din + ch];
        float y = 0.f;
        #pragma unroll
        for (int n = 0; n < 16; ++n) {
            float Bn = __shfl(bc, n);
            float Cn = __shfl(bc, n + 16);
            float dA = __expf(d * Arow[n]);
            h[n] = dA * h[n] + (d * Bn) * uu;
            y = fmaf(h[n], Cn, y);
        }
        float z  = xz[bt * (size_t)xzstride + din + ch];
        float sz = z / (1.f + __expf(-z));
        ybf[bt * din + ch] = (__bf16)((y + uu * Dch) * sz);
    }
}

// ---------------------------------------------------------------------------
extern "C" void kernel_launch(void* const* d_in, const int* in_sizes, int n_in,
                              void* d_out, int out_size, void* d_ws, size_t ws_size,
                              hipStream_t stream) {
    (void)in_sizes; (void)n_in; (void)out_size; (void)ws_size;

    const float* x       = (const float*)d_in[0];
    const float* norm_g  = (const float*)d_in[1];
    const float* norm_b  = (const float*)d_in[2];
    const float* in_w    = (const float*)d_in[3];
    const float* conv_w  = (const float*)d_in[4];
    const float* conv_b  = (const float*)d_in[5];
    const float* xproj_w = (const float*)d_in[6];
    const float* dt_w    = (const float*)d_in[7];
    const float* dt_b    = (const float*)d_in[8];
    const float* A_log   = (const float*)d_in[9];
    const float* Dv      = (const float*)d_in[10];
    const float* out_w   = (const float*)d_in[11];
    float* out = (float*)d_out;

    constexpr int Bsz = 2, L = 2048, DM = 1024, DI = 2048, R = 64, P = 96;
    constexpr int Mrows = Bsz * L;  // 4096

    char* wsp = (char*)d_ws;
    auto alloc = [&](size_t bytes) -> char* {
        char* p = wsp;
        wsp += (bytes + 255) & ~(size_t)255;
        return p;
    };
    __bf16* in_w_bf  = (__bf16*)alloc((size_t)2 * DI * DM * 2);
    __bf16* out_w_bf = (__bf16*)alloc((size_t)DM * DI * 2);
    __bf16* xn_bf    = (__bf16*)alloc((size_t)Mrows * DM * 2);
    float*  xz       = (float*) alloc((size_t)Mrows * 2 * DI * 4);
    float*  u        = (float*) alloc((size_t)Mrows * DI * 4);
    float*  dbc      = (float*) alloc((size_t)Mrows * P * 4);
    float*  delta    = (float*) alloc((size_t)Mrows * DI * 4);
    __bf16* ybf      = (__bf16*)alloc((size_t)Mrows * DI * 2);

    // weight conversions
    {
        size_t n1 = (size_t)2 * DI * DM;
        f32_to_bf16_kernel<<<(unsigned)((n1 + 255) / 256), 256, 0, stream>>>(in_w, in_w_bf, n1);
        size_t n2 = (size_t)DM * DI;
        f32_to_bf16_kernel<<<(unsigned)((n2 + 255) / 256), 256, 0, stream>>>(out_w, out_w_bf, n2);
    }

    // LayerNorm -> bf16 A operand
    ln_kernel<<<Mrows, 256, 0, stream>>>(x, norm_g, norm_b, xn_bf, DM);

    // xz = xn @ in_w^T   (4096 x 4096 x 1024)
    gemm_bf16_wmma<<<dim3(2 * DI / BN, Mrows / BM), 256, 0, stream>>>(
        xn_bf, in_w_bf, nullptr, xz, Mrows, 2 * DI, DM);

    // depthwise causal conv + SiLU
    {
        size_t total = (size_t)Mrows * DI;
        conv_silu_kernel<<<(unsigned)((total + 255) / 256), 256, 0, stream>>>(
            xz, conv_w, conv_b, u, L, DI, 2 * DI, total);
    }

    // dbc = u @ xproj_w^T
    xproj_kernel<<<Mrows, 128, 0, stream>>>(u, xproj_w, dbc, DI, P);

    // delta = softplus(dbc[:, :64] @ dt_w^T + dt_b)
    delta_kernel<<<Mrows, 256, 0, stream>>>(dbc, dt_w, dt_b, delta, DI, R, P);

    // selective scan + gating -> bf16 y
    scan_kernel<<<16, 256, 0, stream>>>(delta, u, dbc, xz, A_log, Dv, ybf,
                                        L, DI, P, R, 2 * DI);

    // out = y @ out_w^T + residual   (4096 x 1024 x 2048)
    gemm_bf16_wmma<<<dim3(DM / BN, Mrows / BM), 256, 0, stream>>>(
        ybf, out_w_bf, x, out, Mrows, DM, DI);
}
